// MSCAAttention5_57183194579387
// MI455X (gfx1250) — matrix operand order repaired and checked by reading, not compiled, not run on hardware
//
#include <hip/hip_runtime.h>
#include <cstdint>

// ---------------------------------------------------------------------------
// MSCA attention for MI455X (gfx1250, wave32, WMMA, async global->LDS).
//   K0: convert w3 (512x512 f32) -> bf16 in workspace
//   K1: fused 11-conv depthwise chain, one (b,c) 56x56 plane per workgroup,
//       x plane copied in via GLOBAL_LOAD_ASYNC_TO_LDS_B128; all
//       intermediates stay in LDS; writes attn_sum as bf16
//   K2: per-batch GEMM D = W3 x attn_sum via v_wmma_f32_16x16x32_bf16 with
//       double-buffered async-LDS staging (ASYNCcnt pipelined, last
//       iteration peeled so the steady-state loop is branch-free); fused
//       bias + elementwise (* x) epilogue -> d_out (f32)
// ---------------------------------------------------------------------------

#define HWD 56
#define NPIX 3136          // 56*56
#define CCH 512
#define NB 8

typedef __attribute__((ext_vector_type(16))) __bf16 v16bf;
typedef __attribute__((ext_vector_type(8)))  float  v8f;

__device__ __forceinline__ unsigned short f32_to_bf16(float f) {
  unsigned int u = __float_as_uint(f);
  u = (u + 0x7FFFu + ((u >> 16) & 1u)) >> 16;   // round-to-nearest-even
  return (unsigned short)u;
}

// one wave-level async copy: 16 bytes/lane, global -> LDS (ASYNCcnt++)
__device__ __forceinline__ void async_copy_b128(unsigned lds_off,
                                                const void* gptr) {
  unsigned long long g = (unsigned long long)(uintptr_t)gptr;
  asm volatile("global_load_async_to_lds_b128 %0, %1, off"
               :: "v"(lds_off), "v"(g) : "memory");
}
__device__ __forceinline__ void async_copy_b128_o16(unsigned lds_off,
                                                    const void* gptr) {
  // INST_OFFSET is added to BOTH the LDS and global address (ISA 10.7)
  unsigned long long g = (unsigned long long)(uintptr_t)gptr;
  asm volatile("global_load_async_to_lds_b128 %0, %1, off offset:16"
               :: "v"(lds_off), "v"(g) : "memory");
}
// truncating a generic pointer to a __shared__ object yields the LDS byte
// offset (flat aperture keeps LDS offset in addr[31:0], ISA 10.2)
__device__ __forceinline__ unsigned lds_off_of(const void* p) {
  return (unsigned)(uintptr_t)p;
}

// ---------------------------------------------------------------------------
// K0: weight conversion
// ---------------------------------------------------------------------------
__global__ __launch_bounds__(256) void cvt_w3_kernel(
    const float* __restrict__ w3, unsigned short* __restrict__ dst) {
  int i = blockIdx.x * 256 + threadIdx.x;   // grid covers 512*512 exactly
  dst[i] = f32_to_bf16(w3[i]);
}

// ---------------------------------------------------------------------------
// K1 helpers: separable conv stages inside LDS (implicit zero padding).
// All dims are compile-time so index math folds to const mul/shift.
// ---------------------------------------------------------------------------
template <int K, int DIL, int Hs, int Ws, int Wd>
__device__ __forceinline__ void rowConv(const float* __restrict__ src,
                                        float* __restrict__ dst,
                                        const float* __restrict__ w, float bias,
                                        int pad, int tid) {
  constexpr int n = Hs * Wd;
  for (int i = tid; i < n; i += 256) {
    int y = i / Wd;
    int xo = i - y * Wd;
    const float* s = src + y * Ws;
    float acc = bias;
#pragma unroll
    for (int j = 0; j < K; ++j) {
      int xi = xo + j * DIL - pad;
      if ((unsigned)xi < (unsigned)Ws) acc = __builtin_fmaf(w[j], s[xi], acc);
    }
    dst[i] = acc;
  }
}

// dst[i] = (HASBASE ? base[i] : 0) + colconv(src)[i]
template <int K, int DIL, bool HASBASE, int Hs, int Hd, int W>
__device__ __forceinline__ void colConv(const float* __restrict__ src,
                                        float* __restrict__ dst,
                                        const float* __restrict__ base,
                                        const float* __restrict__ w, float bias,
                                        int pad, int tid) {
  constexpr int n = Hd * W;
  for (int i = tid; i < n; i += 256) {
    int y = i / W;
    int x = i - y * W;
    float acc = bias;
#pragma unroll
    for (int j = 0; j < K; ++j) {
      int yi = y + j * DIL - pad;
      if ((unsigned)yi < (unsigned)Hs) acc = __builtin_fmaf(w[j], src[yi * W + x], acc);
    }
    dst[i] = (HASBASE ? base[i] : 0.0f) + acc;
  }
}

// ---------------------------------------------------------------------------
// K1: fused depthwise chain.  One workgroup (256 thr = 8 waves) per (b,c).
// LDS layout (floats):  A[3136] | S[3136] | T1[3472] | T2X[3844]   (~53 KB)
// ---------------------------------------------------------------------------
__global__ __launch_bounds__(256) void msca_dw_kernel(
    const float* __restrict__ x,
    const float* __restrict__ w0,   const float* __restrict__ b0,
    const float* __restrict__ w0_1, const float* __restrict__ b0_1,
    const float* __restrict__ w0_2, const float* __restrict__ b0_2,
    const float* __restrict__ w1_1, const float* __restrict__ b1_1,
    const float* __restrict__ w1_2, const float* __restrict__ b1_2,
    const float* __restrict__ w1_3, const float* __restrict__ b1_3,
    const float* __restrict__ w1_4, const float* __restrict__ b1_4,
    const float* __restrict__ w2_1, const float* __restrict__ b2_1,
    const float* __restrict__ w2_2, const float* __restrict__ b2_2,
    const float* __restrict__ w2_3, const float* __restrict__ b2_3,
    const float* __restrict__ w2_4, const float* __restrict__ b2_4,
    unsigned short* __restrict__ attn_bf) {
  __shared__ float lds[3136 + 3136 + 3472 + 3844];
  __shared__ float wb[86];
  float* A   = lds;
  float* S   = lds + 3136;
  float* T1  = lds + 6272;
  float* T2X = lds + 9744;   // x plane first, later 62x62 intermediates

  const int tid = threadIdx.x;
  const int plane = blockIdx.x;          // b*512 + c
  const int c = plane & (CCH - 1);

  // async-copy input plane into LDS (pure copy: 784 x 16B chunks)
  const float* xp = x + (size_t)plane * NPIX;
  {
    const unsigned base = lds_off_of(T2X);
    for (int i = tid; i < NPIX / 4; i += 256)
      async_copy_b128(base + i * 16, xp + i * 4);
    asm volatile("s_wait_asynccnt 0" ::: "memory");
  }

  // gather all per-channel weights/biases into LDS (75 taps + 11 biases)
  if (tid < 86) {
    float v;
    int t = tid;
    if      (t < 25)  v = w0[c * 25 + t];
    else if (t == 25) v = b0[c];
    else if (t < 33)  v = w0_1[c * 7 + (t - 26)];
    else if (t == 33) v = b0_1[c];
    else if (t < 41)  v = w0_2[c * 7 + (t - 34)];
    else if (t == 41) v = b0_2[c];
    else if (t < 45)  v = w1_1[c * 3 + (t - 42)];
    else if (t == 45) v = b1_1[c];
    else if (t < 49)  v = w1_2[c * 3 + (t - 46)];
    else if (t == 49) v = b1_2[c];
    else if (t < 55)  v = w1_3[c * 5 + (t - 50)];
    else if (t == 55) v = b1_3[c];
    else if (t < 61)  v = w1_4[c * 5 + (t - 56)];
    else if (t == 61) v = b1_4[c];
    else if (t < 67)  v = w2_1[c * 5 + (t - 62)];
    else if (t == 67) v = b2_1[c];
    else if (t < 73)  v = w2_2[c * 5 + (t - 68)];
    else if (t == 73) v = b2_2[c];
    else if (t < 79)  v = w2_3[c * 5 + (t - 74)];
    else if (t == 79) v = b2_3[c];
    else if (t < 85)  v = w2_4[c * 5 + (t - 80)];
    else              v = b2_4[c];
    wb[t] = v;
  }
  __syncthreads();

  // attn = dw5x5(x) + b0   -> A
  {
    const float* w = wb;
    const float bias = wb[25];
    for (int i = tid; i < NPIX; i += 256) {
      int y = i / HWD;
      int xo = i - y * HWD;
      float acc = bias;
#pragma unroll
      for (int dy = 0; dy < 5; ++dy) {
        int yi = y + dy - 2;
        if ((unsigned)yi < (unsigned)HWD) {
          const float* row = T2X + yi * HWD;
#pragma unroll
          for (int dx = 0; dx < 5; ++dx) {
            int xi = xo + dx - 2;
            if ((unsigned)xi < (unsigned)HWD)
              acc = __builtin_fmaf(w[dy * 5 + dx], row[xi], acc);
          }
        }
      }
      A[i] = acc;
    }
  }
  __syncthreads();

  // branch 0: 1x7 pad3, 7x1 pad3;  S = A + a0
  rowConv<7, 1, 56, 56, 56>(A, T1, wb + 26, wb[33], 3, tid);
  __syncthreads();
  colConv<7, 1, true, 56, 56, 56>(T1, S, A, wb + 34, wb[41], 3, tid);
  __syncthreads();

  // branch 1: 1x3 pad2 (56x58) -> 3x1 pad2 (58x58) -> 1x5 dil2 pad3 (58x56)
  //           -> 5x1 dil2 pad3 (56x56);  S += a1
  rowConv<3, 1, 56, 56, 58>(A, T1, wb + 42, wb[45], 2, tid);
  __syncthreads();
  colConv<3, 1, false, 56, 58, 58>(T1, T2X, nullptr, wb + 46, wb[49], 2, tid);
  __syncthreads();
  rowConv<5, 2, 58, 58, 56>(T2X, T1, wb + 50, wb[55], 3, tid);
  __syncthreads();
  colConv<5, 2, true, 58, 56, 56>(T1, S, S, wb + 56, wb[61], 3, tid);
  __syncthreads();

  // branch 2: 1x5 pad5 (56x62) -> 5x1 pad5 (62x62) -> 1x5 dil4 pad5 (62x56)
  //           -> 5x1 dil4 pad5 (56x56);  S += a2
  rowConv<5, 1, 56, 56, 62>(A, T1, wb + 62, wb[67], 5, tid);
  __syncthreads();
  colConv<5, 1, false, 56, 62, 62>(T1, T2X, nullptr, wb + 68, wb[73], 5, tid);
  __syncthreads();
  rowConv<5, 4, 62, 62, 56>(T2X, T1, wb + 74, wb[79], 5, tid);
  __syncthreads();
  colConv<5, 4, true, 62, 56, 56>(T1, S, S, wb + 80, wb[85], 5, tid);
  __syncthreads();

  // write attn_sum as bf16
  unsigned short* op = attn_bf + (size_t)plane * NPIX;
  for (int i = tid; i < NPIX; i += 256) op[i] = f32_to_bf16(S[i]);
}

// ---------------------------------------------------------------------------
// K2: per-batch GEMM  out[b,co,hw] = (sum_ci W3[co,ci]*attn[b,ci,hw] + b3[co]) * x[b,co,hw]
// Block tile 128(co) x 128(hw), K-step 32, 8 waves -> each wave 64x32
// (4x2 grid of v_wmma_f32_16x16x32_bf16).  LDS staging is double-buffered
// GLOBAL_LOAD_ASYNC_TO_LDS_B128; last K-step peeled so the steady-state
// loop is: issue-next / s_wait_asynccnt 4 / barrier / frags+WMMA / barrier.
// ---------------------------------------------------------------------------
__global__ __launch_bounds__(256) void msca_gemm_kernel(
    const unsigned short* __restrict__ w3bf,    // [512][512] bf16
    const unsigned short* __restrict__ attn_bf, // [8][512][3136] bf16
    const float* __restrict__ b3,
    const float* __restrict__ x,
    float* __restrict__ out) {
  constexpr int LDW = 40;    // ushorts per W row  (32 + 8 pad)
  constexpr int LDA = 136;   // ushorts per A row  (128 + 8 pad)
  constexpr int KT  = CCH / 32;
  __shared__ unsigned short lw[2][128 * LDW];
  __shared__ unsigned short la[2][32 * LDA];

  const int tid  = threadIdx.x;
  const int lane = tid & 31;
  const int wave = tid >> 5;
  const int wco  = wave >> 2;          // 0..1
  const int whw  = wave & 3;           // 0..3
  const int hw0  = blockIdx.x * 128;
  const int co0  = blockIdx.y * 128;
  const int b    = blockIdx.z;

  union Acc { v8f v; float f[8]; };
  Acc acc[4][2];
#pragma unroll
  for (int m = 0; m < 4; ++m)
#pragma unroll
    for (int n = 0; n < 2; ++n)
#pragma unroll
      for (int v = 0; v < 8; ++v) acc[m][n].f[v] = 0.0f;

  // staging assignment: each thread owns 32B of W and 32B of A per K-step
  const int wrow = tid >> 1, wseg = tid & 1;   // W: 128 rows x 2 16-elem segs
  const int arow = tid >> 3, aseg = tid & 7;   // A: 32 rows x 8 16-elem segs
  const unsigned short* wsrc = w3bf + (size_t)(co0 + wrow) * CCH + wseg * 16;
  // clamp OOB hw (only last hw tile): aliased columns are never stored
  int ahw = hw0 + aseg * 16;
  if (ahw > NPIX - 16) ahw = NPIX - 16;
  const unsigned short* asrc = attn_bf + ((size_t)(b * CCH + arow)) * NPIX + ahw;

  unsigned lwoff[2], laoff[2];
#pragma unroll
  for (int p = 0; p < 2; ++p) {
    lwoff[p] = lds_off_of(&lw[p][wrow * LDW + wseg * 16]);
    laoff[p] = lds_off_of(&la[p][arow * LDA + aseg * 16]);
  }

  auto issue_tile = [&](int buf, int kt) {
    const unsigned short* gw = wsrc + kt * 32;            // +64B per K-step
    const unsigned short* ga = asrc + (size_t)kt * 32 * NPIX;
    async_copy_b128(lwoff[buf], gw);
    async_copy_b128_o16(lwoff[buf], gw);                  // both sides +16B
    async_copy_b128(laoff[buf], ga);
    async_copy_b128_o16(laoff[buf], ga);
  };

  // fragment loads (ISA 7.12.2 wave32 layouts) + 8 WMMAs on buffer `cur`
  auto compute_tile = [&](int cur) {
    const unsigned short* lwc = lw[cur];
    const unsigned short* lac = la[cur];
    union FragU { v16bf v; unsigned int u[8]; };
    FragU fw[4], fb[2];
    const int rbase  = wco * 64 + (lane & 15);
    const int koff8  = (lane & 16) ? 8 : 0;
#pragma unroll
    for (int m = 0; m < 4; ++m) {
      const unsigned short* p = &lwc[(rbase + m * 16) * LDW];
#pragma unroll
      for (int v = 0; v < 8; ++v) {
        const int k = ((v & 4) << 2) + koff8 + 2 * (v & 3); // A-frag K map
        fw[m].u[v] = *(const unsigned int*)(p + k);
      }
    }
    const int cbase  = whw * 32 + (lane & 15);
    const int koff16 = (lane & 16) ? 8 : 0;
#pragma unroll
    for (int n = 0; n < 2; ++n) {
      const int col = cbase + n * 16;
#pragma unroll
      for (int v = 0; v < 8; ++v) {
        unsigned int lo = lac[(((v & 4) << 2) + koff16 + 2 * (v & 3)) * LDA + col];
        unsigned int hi = lac[(((v & 4) << 2) + koff16 + 2 * (v & 3) + 1) * LDA + col];
        fb[n].u[v] = lo | (hi << 16);
      }
    }
#pragma unroll
    for (int m = 0; m < 4; ++m)
#pragma unroll
      for (int n = 0; n < 2; ++n)
        acc[m][n].v = __builtin_amdgcn_wmma_f32_16x16x32_bf16(
            false, fw[m].v, false, fb[n].v, (short)0, acc[m][n].v,
            false, false);
  };

  issue_tile(0, 0);

  // steady state: branch-free issue/wait/compute, 15 iterations
#pragma unroll 1
  for (int kt = 0; kt < KT - 1; ++kt) {
    const int cur = kt & 1;
    issue_tile(cur ^ 1, kt + 1);     // prefetch next tile into other buffer
    asm volatile("s_wait_asynccnt 4" ::: "memory"); // in-order => cur landed
    __syncthreads();                 // cur buffers valid across all waves
    compute_tile(cur);
    __syncthreads();                 // frag reads done before refill of cur
  }
  // peeled last K-step
  asm volatile("s_wait_asynccnt 0" ::: "memory");
  __syncthreads();
  compute_tile((KT - 1) & 1);

  // epilogue: bias + multiply by u = x, predicated stores only
  const int mrow = (lane & 16) ? 8 : 0;   // D: M = v + 8*(lane>=16)
#pragma unroll
  for (int m = 0; m < 4; ++m) {
    const int coBase = co0 + wco * 64 + m * 16 + mrow;
#pragma unroll
    for (int n = 0; n < 2; ++n) {
      const int hw = hw0 + whw * 32 + n * 16 + (lane & 15);
      if (hw < NPIX) {
#pragma unroll
        for (int v = 0; v < 8; ++v) {
          const int co = coBase + v;
          const size_t idx = ((size_t)(b * CCH + co)) * NPIX + hw;
          out[idx] = (acc[m][n].f[v] + b3[co]) * x[idx];
        }
      }
    }
  }
}

// ---------------------------------------------------------------------------
// Launch. Workspace layout: [0, 512KB) w3 bf16 | [512KB, +25.7MB) attn bf16.
// ---------------------------------------------------------------------------
extern "C" void kernel_launch(void* const* d_in, const int* in_sizes, int n_in,
                              void* d_out, int out_size, void* d_ws, size_t ws_size,
                              hipStream_t stream) {
  (void)in_sizes; (void)n_in; (void)out_size; (void)ws_size;
  const float* x    = (const float*)d_in[0];
  const float* w0   = (const float*)d_in[1];
  const float* b0   = (const float*)d_in[2];
  const float* w0_1 = (const float*)d_in[3];
  const float* b0_1 = (const float*)d_in[4];
  const float* w0_2 = (const float*)d_in[5];
  const float* b0_2 = (const float*)d_in[6];
  const float* w1_1 = (const float*)d_in[7];
  const float* b1_1 = (const float*)d_in[8];
  const float* w1_2 = (const float*)d_in[9];
  const float* b1_2 = (const float*)d_in[10];
  const float* w1_3 = (const float*)d_in[11];
  const float* b1_3 = (const float*)d_in[12];
  const float* w1_4 = (const float*)d_in[13];
  const float* b1_4 = (const float*)d_in[14];
  const float* w2_1 = (const float*)d_in[15];
  const float* b2_1 = (const float*)d_in[16];
  const float* w2_2 = (const float*)d_in[17];
  const float* b2_2 = (const float*)d_in[18];
  const float* w2_3 = (const float*)d_in[19];
  const float* b2_3 = (const float*)d_in[20];
  const float* w2_4 = (const float*)d_in[21];
  const float* b2_4 = (const float*)d_in[22];
  const float* w3   = (const float*)d_in[23];
  const float* b3   = (const float*)d_in[24];
  float* out = (float*)d_out;

  unsigned short* w3bf   = (unsigned short*)d_ws;                    // 512KB
  unsigned short* attnbf = w3bf + (size_t)CCH * CCH;                 // 25.7MB

  cvt_w3_kernel<<<(CCH * CCH) / 256, 256, 0, stream>>>(w3, w3bf);

  msca_dw_kernel<<<NB * CCH, 256, 0, stream>>>(
      x, w0, b0, w0_1, b0_1, w0_2, b0_2,
      w1_1, b1_1, w1_2, b1_2, w1_3, b1_3, w1_4, b1_4,
      w2_1, b2_1, w2_2, b2_2, w2_3, b2_3, w2_4, b2_4, attnbf);

  dim3 grid((NPIX + 127) / 128, CCH / 128, NB);   // 25 x 4 x 8
  msca_gemm_kernel<<<grid, 256, 0, stream>>>(w3bf, attnbf, b3, x, out);
}
